// FP8DynamicLinear_31344671326525
// MI455X (gfx1250) — 8-bit matrix operand order repaired and checked by reading, compiled, not run on hardware
//
#include <hip/hip_runtime.h>
#include <stdint.h>

typedef __attribute__((ext_vector_type(16))) int   v16i;
typedef __attribute__((ext_vector_type(8)))  float v8f;
typedef __attribute__((ext_vector_type(4)))  int   v4i;

#define E4M3_AMAX 216.0f
#define FP8_CLIP  240.0f

#define TILE_K  128
#define APITCH  144
#define BLK_M   128
#define BLK_N   64
#define A_BYTES (BLK_M * APITCH)
#define B_BYTES (BLK_N * APITCH)
#define BUF_BYTES (A_BYTES + B_BYTES)

__device__ __forceinline__ unsigned umax_(unsigned a, unsigned b) { return a > b ? a : b; }

__device__ __forceinline__ void async_cp16(const void* g, void* l) {
#if __has_builtin(__builtin_amdgcn_global_load_async_to_lds_b128)
  __builtin_amdgcn_global_load_async_to_lds_b128(
      (__attribute__((address_space(1))) v4i*)g,
      (__attribute__((address_space(3))) v4i*)l, 0, 0);
#else
  *(uint4*)l = *(const uint4*)g;
#endif
}

__device__ __forceinline__ void wait_async_all() {
#if __has_builtin(__builtin_amdgcn_s_wait_asynccnt)
  __builtin_amdgcn_s_wait_asynccnt(0);
#else
  asm volatile("s_wait_asynccnt 0x0" ::: "memory");
#endif
}

__device__ __forceinline__ unsigned char f32_to_e4m3_sw(float x) {
  unsigned b    = __float_as_uint(x);
  unsigned sign = (b >> 24) & 0x80u;
  float ax = fabsf(x);
  if (ax < 0.015625f) {
    int q = (int)rintf(ax * 512.0f);
    if (q >= 8) return (unsigned char)(sign | 0x08u);
    return (unsigned char)(sign | (unsigned)q);
  }
  unsigned ab  = b & 0x7FFFFFFFu;
  unsigned lsb = (ab >> 20) & 1u;
  ab += 0x0007FFFFu + lsb;
  int e = (int)(ab >> 23) - 127;
  unsigned man = (ab >> 20) & 7u;
  int ef = e + 7;
  if (ef > 15) { ef = 15; man = 6u; }
  return (unsigned char)(sign | ((unsigned)ef << 3) | man);
}

__global__ void init_amax_kernel(unsigned* a) { a[0] = 0u; a[1] = 0u; }

__global__ __launch_bounds__(256)
void amax_kernel(const float* __restrict__ x, long n, unsigned* __restrict__ out) {
  __shared__ unsigned sred[256];
  unsigned m = 0;
  const long stride = (long)gridDim.x * 256 * 4;
  for (long i = ((long)blockIdx.x * 256 + threadIdx.x) * 4; i < n; i += stride) {
    float4 v = *(const float4*)(x + i);
    m = umax_(m, __float_as_uint(fabsf(v.x)));
    m = umax_(m, __float_as_uint(fabsf(v.y)));
    m = umax_(m, __float_as_uint(fabsf(v.z)));
    m = umax_(m, __float_as_uint(fabsf(v.w)));
  }
  sred[threadIdx.x] = m;
  __syncthreads();
  for (int s = 128; s > 0; s >>= 1) {
    if ((int)threadIdx.x < s) sred[threadIdx.x] = umax_(sred[threadIdx.x], sred[threadIdx.x + s]);
    __syncthreads();
  }
  if (threadIdx.x == 0) atomicMax(out, sred[0]);
}

__global__ __launch_bounds__(256)
void quant_fp8_kernel(const float* __restrict__ x, unsigned char* __restrict__ q,
                      const unsigned* __restrict__ amaxb, int idx, long n) {
  long i = ((long)blockIdx.x * 256 + threadIdx.x) * 4;
  if (i >= n) return;
  const float s = E4M3_AMAX / __uint_as_float(amaxb[idx]);
  float4 v = *(const float4*)(x + i);
  float a = fminf(fmaxf(v.x * s, -FP8_CLIP), FP8_CLIP);
  float b = fminf(fmaxf(v.y * s, -FP8_CLIP), FP8_CLIP);
  float c = fminf(fmaxf(v.z * s, -FP8_CLIP), FP8_CLIP);
  float d = fminf(fmaxf(v.w * s, -FP8_CLIP), FP8_CLIP);
  unsigned packed;
#if __has_builtin(__builtin_amdgcn_cvt_pk_fp8_f32)
  int p = 0;
  p = __builtin_amdgcn_cvt_pk_fp8_f32(a, b, p, false);
  p = __builtin_amdgcn_cvt_pk_fp8_f32(c, d, p, true);
  packed = (unsigned)p;
#else
  packed = (unsigned)f32_to_e4m3_sw(a)        | ((unsigned)f32_to_e4m3_sw(b) << 8) |
           ((unsigned)f32_to_e4m3_sw(c) << 16) | ((unsigned)f32_to_e4m3_sw(d) << 24);
#endif
  *(unsigned*)(q + i) = packed;
}

__global__ __launch_bounds__(256)
void gemm_fp8_kernel(const unsigned char* __restrict__ Xq,
                     const unsigned char* __restrict__ Wq,
                     const float* __restrict__ bias,
                     const unsigned* __restrict__ amaxb,
                     float* __restrict__ out, int K, int N) {
  __shared__ __align__(16) unsigned char smem[2 * BUF_BYTES];

  const int tid  = threadIdx.x;
  const int lane = tid & 31;
  const int wv   = tid >> 5;
  const int wm   = wv & 3;
  const int wn   = wv >> 2;
  const int l15  = lane & 15;
  const int hsel = lane >> 4;

  const long m0 = (long)blockIdx.y * BLK_M;
  const int  n0 = blockIdx.x * BLK_N;
  const int  kTiles = K / TILE_K;

  auto stage = [&](int buf, int k0) {
    unsigned char* base = smem + buf * BUF_BYTES;
#pragma unroll
    for (int r = 0; r < 4; ++r) {
      int c = tid + r * 256;
      int row = c >> 3, ch = (c & 7) * 16;
      async_cp16(Xq + (m0 + row) * (long)K + k0 + ch, base + row * APITCH + ch);
    }
#pragma unroll
    for (int r = 0; r < 2; ++r) {
      int c = tid + r * 256;
      int row = c >> 3, ch = (c & 7) * 16;
      async_cp16(Wq + (long)(n0 + row) * K + k0 + ch, base + A_BYTES + row * APITCH + ch);
    }
  };

  v8f acc00 = {0,0,0,0,0,0,0,0};
  v8f acc01 = acc00, acc10 = acc00, acc11 = acc00;

  stage(0, 0);
  wait_async_all();
  __syncthreads();

  for (int kt = 0; kt < kTiles; ++kt) {
    const int cur = kt & 1;
    if (kt + 1 < kTiles) stage(cur ^ 1, (kt + 1) * TILE_K);

    const unsigned char* aB = smem + cur * BUF_BYTES;
    const unsigned char* bB = aB + A_BYTES;

    v16i a0, a1;
    const unsigned char* ap0 = aB + (wm * 32 + l15) * APITCH + hsel * 16;
    const unsigned char* ap1 = ap0 + 16 * APITCH;
#pragma unroll
    for (int i = 0; i < 4; ++i) {
      uint4 d0 = *(const uint4*)(ap0 + i * 32);
      uint4 d1 = *(const uint4*)(ap1 + i * 32);
      a0[4*i] = (int)d0.x; a0[4*i+1] = (int)d0.y; a0[4*i+2] = (int)d0.z; a0[4*i+3] = (int)d0.w;
      a1[4*i] = (int)d1.x; a1[4*i+1] = (int)d1.y; a1[4*i+2] = (int)d1.z; a1[4*i+3] = (int)d1.w;
    }
    v16i b0, b1;
    const unsigned char* bp0 = bB + (wn * 32 + l15) * APITCH + hsel * 16;
    const unsigned char* bp1 = bp0 + 16 * APITCH;
#pragma unroll
    for (int j = 0; j < 4; ++j) {
      uint4 d0 = *(const uint4*)(bp0 + j * 32);
      uint4 d1 = *(const uint4*)(bp1 + j * 32);
      b0[4*j]=(int)d0.x; b0[4*j+1]=(int)d0.y; b0[4*j+2]=(int)d0.z; b0[4*j+3]=(int)d0.w;
      b1[4*j]=(int)d1.x; b1[4*j+1]=(int)d1.y; b1[4*j+2]=(int)d1.z; b1[4*j+3]=(int)d1.w;
    }

    acc00 = __builtin_amdgcn_wmma_f32_16x16x128_fp8_fp8(a0, b0, (short)0, acc00, false, false);
    acc01 = __builtin_amdgcn_wmma_f32_16x16x128_fp8_fp8(a0, b1, (short)0, acc01, false, false);
    acc10 = __builtin_amdgcn_wmma_f32_16x16x128_fp8_fp8(a1, b0, (short)0, acc10, false, false);
    acc11 = __builtin_amdgcn_wmma_f32_16x16x128_fp8_fp8(a1, b1, (short)0, acc11, false, false);

    wait_async_all();
    __syncthreads();
  }

  const float dq = (__uint_as_float(amaxb[0]) * (1.0f / E4M3_AMAX)) *
                   (__uint_as_float(amaxb[1]) * (1.0f / E4M3_AMAX));
  const int  col0 = n0 + wn * 32 + l15;
  const long rb0  = m0 + wm * 32 + hsel * 8;
  const float bv0 = bias[col0];
  const float bv1 = bias[col0 + 16];
#pragma unroll
  for (int r = 0; r < 8; ++r) {
    long row = rb0 + r;
    out[row * N + col0]             = acc00[r] * dq + bv0;
    out[row * N + col0 + 16]        = acc01[r] * dq + bv1;
    out[(row + 16) * N + col0]      = acc10[r] * dq + bv0;
    out[(row + 16) * N + col0 + 16] = acc11[r] * dq + bv1;
  }
}

extern "C" void kernel_launch(void* const* d_in, const int* in_sizes, int n_in,
                              void* d_out, int out_size, void* d_ws, size_t ws_size,
                              hipStream_t stream) {
  (void)n_in; (void)out_size; (void)ws_size;
  const float* inp    = (const float*)d_in[0];
  const float* weight = (const float*)d_in[1];
  const float* bias   = (const float*)d_in[2];
  float* out = (float*)d_out;

  const long n_inp  = (long)in_sizes[0];
  const long n_w    = (long)in_sizes[1];
  const int  N      = in_sizes[2];
  const int  K      = (int)(n_w / N);
  const long tokens = n_inp / K;

  unsigned*      amaxb = (unsigned*)d_ws;
  unsigned char* x8    = (unsigned char*)d_ws + 256;
  unsigned char* w8    = x8 + n_inp;

  init_amax_kernel<<<1, 1, 0, stream>>>(amaxb);

  amax_kernel<<<2048, 256, 0, stream>>>(inp,    n_inp, amaxb + 0);
  amax_kernel<<<1024, 256, 0, stream>>>(weight, n_w,   amaxb + 1);

  quant_fp8_kernel<<<(unsigned)((n_inp / 4 + 255) / 256), 256, 0, stream>>>(inp,    x8, amaxb, 0, n_inp);
  quant_fp8_kernel<<<(unsigned)((n_w   / 4 + 255) / 256), 256, 0, stream>>>(weight, w8, amaxb, 1, n_w);

  dim3 grid((unsigned)(N / BLK_N), (unsigned)(tokens / BLK_M));
  gemm_fp8_kernel<<<grid, 256, 0, stream>>>(x8, w8, bias, amaxb, out, K, N);
}
